// GraphAttentionLayer_67886253081015
// MI455X (gfx1250) — matrix operand order repaired and checked by reading, not compiled, not run on hardware
//
#include <hip/hip_runtime.h>
#include <hip/hip_bf16.h>

#define NROWS 8192
#define NDIM  256
#define NEG_BIG (-1.0e9f)
#define NEG_BIG_BITS 0xCE6E6B28u
#define SLOPE 0.2f

typedef __attribute__((ext_vector_type(16))) __bf16 v16bf;
typedef __attribute__((ext_vector_type(8)))  float  v8f;

union FragU {
    uint4 q[2];
    v16bf v;
};

__device__ __forceinline__ unsigned short f32_to_bf16(float x) {
    union { float f; unsigned int u; } c; c.f = x;
    unsigned int u = c.u;
    u += 0x7FFFu + ((u >> 16) & 1u);   // round to nearest even
    return (unsigned short)(u >> 16);
}

__device__ __forceinline__ float lrelu(float t) {
    // t>0 -> t ; t<0 -> 0.2*t   (branchless)
    return fmaxf(t, SLOPE * t);
}

// Branchless: adj>0 ? l : -1e9  via bit-select (forces v_cndmask, never exec branch)
__device__ __forceinline__ float sel_masked(int a, float l) {
    unsigned m = (a > 0) ? 0xFFFFFFFFu : 0u;
    unsigned r = (__float_as_uint(l) & m) | (NEG_BIG_BITS & ~m);
    return __uint_as_float(r);
}

// ---------------------------------------------------------------------------
// Kernel 1: Wh = h @ W (f32), f = Wh @ a, and store Wh in bf16 pre-swizzled
// into the WMMA B-fragment layout:
//   tile (tk = K/32, tn = N/16); within tile: lane = (N%16) + 16*((K%32)/16),
//   half = K%16 ; element index = ((tk*16+tn)*32 + lane)*16 + half
// ---------------------------------------------------------------------------
__global__ void __launch_bounds__(256)
wh_kernel(const float* __restrict__ h, const float* __restrict__ W,
          const float* __restrict__ a, float* __restrict__ fvec,
          unsigned short* __restrict__ whStaged)
{
    __shared__ float hs[16 * 256];
    __shared__ float red[8];
    const int t  = threadIdx.x;
    const int i0 = blockIdx.x * 16;

    for (int idx = t; idx < 16 * 256; idx += 256)
        hs[idx] = h[(size_t)i0 * 256 + idx];
    __syncthreads();

    float acc[16];
#pragma unroll
    for (int r = 0; r < 16; ++r) acc[r] = 0.0f;

    const int n = t;  // output column owned by this thread
    for (int k = 0; k < 256; ++k) {
        float wk = W[k * 256 + n];
#pragma unroll
        for (int r = 0; r < 16; ++r)
            acc[r] = fmaf(hs[r * 256 + k], wk, acc[r]);
    }

    // staged bf16 store of Wh (B-fragment layout)
#pragma unroll
    for (int r = 0; r < 16; ++r) {
        int j   = i0 + r;            // K index
        int tk  = j >> 5;
        int hh  = j & 15;
        int lh  = (j >> 4) & 1;
        int tn  = n >> 4;
        int ln  = (n & 15) | (lh << 4);
        whStaged[(((size_t)tk * 16 + tn) * 32 + ln) * 16 + hh] = f32_to_bf16(acc[r]);
    }

    // f[i] = sum_n Wh[i][n] * a[n]
    const float an   = a[n];
    const int   lane = t & 31;
    const int   wave = t >> 5;
    for (int r = 0; r < 16; ++r) {
        float p = acc[r] * an;
#pragma unroll
        for (int off = 16; off; off >>= 1)
            p += __shfl_xor(p, off, 32);
        if (lane == 0) red[wave] = p;
        __syncthreads();
        if (t == 0) {
            float s = 0.0f;
            for (int w = 0; w < 8; ++w) s += red[w];
            fvec[i0 + r] = s;
        }
        __syncthreads();
    }
}

// ---------------------------------------------------------------------------
// Kernel 2: per-row online softmax statistics over masked e.
//   e_masked[i][j] = adj>0 ? leaky_relu(f_i+f_j) : -1e9
//   rowmax[i] = max_j e_masked ; invsum[i] = 1 / sum_j exp(e_masked - rowmax)
// One wave per row (8 rows per 256-thread block). Single adjacency pass.
// Empty rows naturally yield m=-1e9, s=8192 (uniform softmax), matching ref.
// ---------------------------------------------------------------------------
__global__ void __launch_bounds__(256)
rowstat_kernel(const int* __restrict__ adj, const float* __restrict__ fvec,
               float* __restrict__ rowmax, float* __restrict__ invsum)
{
    const int t    = threadIdx.x;
    const int lane = t & 31;
    const int wave = t >> 5;
    const int i    = blockIdx.x * 8 + wave;

    const float fi = fvec[i];
    float m = NEG_BIG, s = 0.0f;

    for (int c = lane * 4; c < NROWS; c += 128) {
        int4   a4 = *(const int4*)(adj + (size_t)i * NROWS + c);
        float4 f4 = *(const float4*)(fvec + c);
#define ONLINE(AV, FV)                                                   \
        { float v  = sel_masked((AV), lrelu(fi + (FV)));                 \
          float nm = fmaxf(m, v);                                        \
          s = s * __expf(m - nm) + __expf(v - nm);                       \
          m = nm; }
        ONLINE(a4.x, f4.x) ONLINE(a4.y, f4.y) ONLINE(a4.z, f4.z) ONLINE(a4.w, f4.w)
#undef ONLINE
    }
#pragma unroll
    for (int off = 16; off; off >>= 1) {
        float m2 = __shfl_xor(m, off, 32);
        float s2 = __shfl_xor(s, off, 32);
        float nm = fmaxf(m, m2);
        s = s * __expf(m - nm) + s2 * __expf(m2 - nm);
        m = nm;
    }
    if (lane == 0) {
        rowmax[i] = m;
        invsum[i] = 1.0f / s;
    }
}

// ---------------------------------------------------------------------------
// Kernel 3: fused alpha + GEMM + elu.
// One WG (256 thr = 8 waves) per 32 output rows. K-chunks of 64 columns:
//   cooperative: adj -> alpha (f32, written to d_out) -> bf16 into LDS A-tile
//   per wave:    2 A-fragments (LDS) x 2 B-fragments (staged Wh, global)
//                -> 8 v_wmma_f32_16x16x32_bf16 per chunk
// Epilogue: elu on the 32x256 f32 accumulators.
// ---------------------------------------------------------------------------
__global__ void __launch_bounds__(256)
gat_fused_kernel(const int* __restrict__ adj, const float* __restrict__ fvec,
                 const float* __restrict__ rowmax, const float* __restrict__ invsum,
                 const unsigned short* __restrict__ whStaged,
                 float* __restrict__ outElu, float* __restrict__ outAlpha)
{
    __shared__ float sf[32], sm[32], sis[32];
    __shared__ __align__(16) unsigned short As[32][72];   // +8 pad: conflict-free

    const int t  = threadIdx.x;
    const int i0 = blockIdx.x * 32;
    if (t < 32) {
        sf[t]  = fvec[i0 + t];
        sm[t]  = rowmax[i0 + t];
        sis[t] = invsum[i0 + t];
    }
    __syncthreads();

    const int lane   = t & 31;
    const int wave   = t >> 5;
    const int laneHi = lane >> 4;
    const int lmod   = lane & 15;
    const int rA     = t >> 4;        // 0..15 (row group for alpha stage)
    const int c4     = (t & 15) * 4;  // col*4 within 64-chunk
    const int tn0    = wave * 2;
    const int tn1    = wave * 2 + 1;

    // Hoist loop-invariant per-row scalars (rows rA and rA+16) into registers.
    const float fi0 = sf[rA],      rm0 = sm[rA],      is0 = sis[rA];
    const float fi1 = sf[rA + 16], rm1 = sm[rA + 16], is1 = sis[rA + 16];

    const uint4* __restrict__ B = (const uint4*)whStaged;

    v8f c00 = {}, c01 = {}, c10 = {}, c11 = {};

    for (int jc = 0; jc < NROWS; jc += 64) {
        // ---- cooperative alpha stage: 32 rows x 64 cols ----
        float4 f4 = *(const float4*)(fvec + jc + c4);
#pragma unroll
        for (int half = 0; half < 2; ++half) {
            const int    r      = rA + half * 16;
            const float  fi     = half ? fi1 : fi0;
            const float  rm     = half ? rm1 : rm0;
            const float  is     = half ? is1 : is0;
            const size_t rowOff = (size_t)(i0 + r) * NROWS + jc + c4;
            int4 a4 = *(const int4*)(adj + rowOff);
            __builtin_prefetch(adj + rowOff + 64, 0, 1);   // next chunk
            float al0, al1, al2, al3;
            { float v = sel_masked(a4.x, lrelu(fi + f4.x)); al0 = __expf(v - rm) * is; }
            { float v = sel_masked(a4.y, lrelu(fi + f4.y)); al1 = __expf(v - rm) * is; }
            { float v = sel_masked(a4.z, lrelu(fi + f4.z)); al2 = __expf(v - rm) * is; }
            { float v = sel_masked(a4.w, lrelu(fi + f4.w)); al3 = __expf(v - rm) * is; }
            *(float4*)(outAlpha + rowOff) = make_float4(al0, al1, al2, al3);
            As[r][c4 + 0] = f32_to_bf16(al0);
            As[r][c4 + 1] = f32_to_bf16(al1);
            As[r][c4 + 2] = f32_to_bf16(al2);
            As[r][c4 + 3] = f32_to_bf16(al3);
        }
        __syncthreads();

        // ---- WMMA stage: two K=32 steps ----
#pragma unroll
        for (int kkI = 0; kkI < 2; ++kkI) {
            const int kk = kkI * 32;
            const int tk = (jc + kk) >> 5;

            // A fragments from LDS (16-bit A 16x32 layout)
            FragU a0u, a1u;
            {
                const unsigned short* ap0 = &As[lmod][kk + laneHi * 16];
                a0u.q[0] = *(const uint4*)(ap0);
                a0u.q[1] = *(const uint4*)(ap0 + 8);
                const unsigned short* ap1 = &As[16 + lmod][kk + laneHi * 16];
                a1u.q[0] = *(const uint4*)(ap1);
                a1u.q[1] = *(const uint4*)(ap1 + 8);
            }
            // B fragments: pre-swizzled, 32B contiguous per lane
            FragU b0u, b1u;
            {
                const uint4* bp0 = B + ((size_t)(tk * 16 + tn0) * 64 + lane * 2);
                b0u.q[0] = bp0[0]; b0u.q[1] = bp0[1];
                const uint4* bp1 = B + ((size_t)(tk * 16 + tn1) * 64 + lane * 2);
                b1u.q[0] = bp1[0]; b1u.q[1] = bp1[1];
            }
            c00 = __builtin_amdgcn_wmma_f32_16x16x32_bf16(false, a0u.v, false, b0u.v, (short)0, c00, false, false);
            c01 = __builtin_amdgcn_wmma_f32_16x16x32_bf16(false, a0u.v, false, b1u.v, (short)0, c01, false, false);
            c10 = __builtin_amdgcn_wmma_f32_16x16x32_bf16(false, a1u.v, false, b0u.v, (short)0, c10, false, false);
            c11 = __builtin_amdgcn_wmma_f32_16x16x32_bf16(false, a1u.v, false, b1u.v, (short)0, c11, false, false);
        }
        __syncthreads();
    }

    // ---- epilogue: elu + store h_prime (16x16 f32 C/D layout) ----
#pragma unroll
    for (int v = 0; v < 8; ++v) {
        const int mrow0 = i0 + v + 8 * laneHi;        // tile mt=0
        const int mrow1 = mrow0 + 16;                 // tile mt=1
        const int gc0   = wave * 32 + lmod;
        const int gc1   = gc0 + 16;
        float x;
        x = c00[v]; outElu[(size_t)mrow0 * NDIM + gc0] = x > 0.0f ? x : __expf(x) - 1.0f;
        x = c01[v]; outElu[(size_t)mrow0 * NDIM + gc1] = x > 0.0f ? x : __expf(x) - 1.0f;
        x = c10[v]; outElu[(size_t)mrow1 * NDIM + gc0] = x > 0.0f ? x : __expf(x) - 1.0f;
        x = c11[v]; outElu[(size_t)mrow1 * NDIM + gc1] = x > 0.0f ? x : __expf(x) - 1.0f;
    }
}

// ---------------------------------------------------------------------------
extern "C" void kernel_launch(void* const* d_in, const int* in_sizes, int n_in,
                              void* d_out, int out_size, void* d_ws, size_t ws_size,
                              hipStream_t stream) {
    const float* h   = (const float*)d_in[0];   // [8192,256]
    const int*   adj = (const int*)  d_in[1];   // [8192,8192]
    const float* W   = (const float*)d_in[2];   // [256,256]
    const float* a   = (const float*)d_in[3];   // [256]

    float* out      = (float*)d_out;
    float* outElu   = out;                              // [8192,256]
    float* outAlpha = out + (size_t)NROWS * NDIM;       // [8192,8192]

    char* ws = (char*)d_ws;
    float*          fvec   = (float*)(ws);               // 32 KB
    float*          rmax   = (float*)(ws + 32768);       // 32 KB
    float*          isum   = (float*)(ws + 65536);       // 32 KB
    unsigned short* staged = (unsigned short*)(ws + 98304); // 4 MB bf16 Wh

    wh_kernel<<<dim3(NROWS / 16), dim3(256), 0, stream>>>(h, W, a, fvec, staged);
    rowstat_kernel<<<dim3(NROWS / 8), dim3(256), 0, stream>>>(adj, fvec, rmax, isum);
    gat_fused_kernel<<<dim3(NROWS / 32), dim3(256), 0, stream>>>(
        adj, fvec, rmax, isum, staged, outElu, outAlpha);
}
